// GIN_74268574482528
// MI455X (gfx1250) — matrix operand order repaired and checked by reading, not compiled, not run on hardware
//
#include <hip/hip_runtime.h>
#include <math.h>

#define HDIM 64

typedef __attribute__((ext_vector_type(16))) __bf16       v16bf;
typedef __attribute__((ext_vector_type(8)))  float        v8f;
typedef __attribute__((ext_vector_type(8)))  unsigned int v8u;

__device__ __forceinline__ unsigned pack2bf(float lo, float hi) {
#if __has_builtin(__builtin_amdgcn_cvt_pk_bf16_f32)
    auto r = __builtin_amdgcn_cvt_pk_bf16_f32(lo, hi);
    return __builtin_bit_cast(unsigned, r);
#else
    __bf16 l = (__bf16)lo, h = (__bf16)hi;
    unsigned short ul = __builtin_bit_cast(unsigned short, l);
    unsigned short uh = __builtin_bit_cast(unsigned short, h);
    return ((unsigned)uh << 16) | (unsigned)ul;
#endif
}

// ---------------- utility kernels ----------------

__global__ void zero_f32(float* p, int n) {
    int i = blockIdx.x * blockDim.x + threadIdx.x;
    if (i < n) p[i] = 0.f;
}

__global__ void copy_f32x4(float4* dst, const float4* src, int n4) {
    int i = blockIdx.x * blockDim.x + threadIdx.x;
    if (i < n4) dst[i] = src[i];
}

// a[dst] += h[src]: one thread per (edge, 4-feature chunk)
__global__ void scatter_add(const float* __restrict__ h, const int* __restrict__ ei,
                            float* __restrict__ a, int E) {
    int i = blockIdx.x * blockDim.x + threadIdx.x;
    int total = E * (HDIM / 4);
    if (i >= total) return;
    int e = i >> 4, q = (i & 15) * 4;
    int s = ei[e];
    int d = ei[E + e];
    float4 v = *(const float4*)(h + (size_t)s * HDIM + q);
    float* ad = a + (size_t)d * HDIM + q;
    atomicAdd(ad + 0, v.x);
    atomicAdd(ad + 1, v.y);
    atomicAdd(ad + 2, v.z);
    atomicAdd(ad + 3, v.w);
}

// g[batch[n]] += h[n]: one thread per (node, 4-feature chunk)
__global__ void pool_add(const float* __restrict__ h, const int* __restrict__ batch,
                         float* __restrict__ g, int N) {
    int i = blockIdx.x * blockDim.x + threadIdx.x;
    int total = N * (HDIM / 4);
    if (i >= total) return;
    int n = i >> 4, q = (i & 15) * 4;
    float4 v = *(const float4*)(h + (size_t)n * HDIM + q);
    float* gd = g + (size_t)batch[n] * HDIM + q;
    atomicAdd(gd + 0, v.x);
    atomicAdd(gd + 1, v.y);
    atomicAdd(gd + 2, v.z);
    atomicAdd(gd + 3, v.w);
}

// stats[0:64]=sum, stats[64:128]=sumsq  ->  ss[0:64]=scale, ss[64:128]=shift
__global__ void bn_finalize(const float* __restrict__ stats, const float* __restrict__ gamma,
                            const float* __restrict__ beta, float* __restrict__ ss, float Nf) {
    int t = threadIdx.x;
    if (t >= HDIM) return;
    float mu  = stats[t] / Nf;
    float var = stats[HDIM + t] / Nf - mu * mu;
    float sc  = gamma[t] * rsqrtf(var + 1e-5f);
    ss[t]        = sc;
    ss[HDIM + t] = beta[t] - mu * sc;
}

// out[r,o] = b2[o] + sum_k relu(gz1[r,k]*scale[k]+shift[k]) * W2[k,o]
__global__ void head_out_kernel(const float* __restrict__ gz1, const float* __restrict__ ss,
                                const float* __restrict__ W2, const float* __restrict__ b2,
                                float* __restrict__ out, int G, int OUT) {
    int i = blockIdx.x * blockDim.x + threadIdx.x;
    if (i >= G * OUT) return;
    int r = i / OUT, o = i - r * OUT;
    float acc = b2[o];
    #pragma unroll
    for (int k = 0; k < HDIM; ++k) {
        float z = fmaxf(fmaf(gz1[(size_t)r * HDIM + k], ss[k], ss[HDIM + k]), 0.f);
        acc = fmaf(z, W2[k * OUT + o], acc);
    }
    out[i] = acc;
}

// ---------------- WMMA GEMM: [N,64] x [64,64] + bias ----------------
// MODE 0: Out = A @ W + bias, accumulate BN stats (sum, sumsq) per column
// MODE 1: Out = gelu( relu(A*scale+shift) @ W + bias )
template <int MODE>
__global__ __launch_bounds__(256) void gin_gemm(
    const float* __restrict__ A, const float* __restrict__ Wg,
    const float* __restrict__ bias, const float* __restrict__ ss,
    float* __restrict__ Out, float* __restrict__ stats, int N)
{
    __shared__ __align__(32) unsigned lwp[2 * 4 * 32 * 8];  // [kkidx][tile][lane][dword]
    __shared__ float lsum[HDIM], lsq[HDIM];
    __shared__ float lss[2 * HDIM];

    int tid = threadIdx.x;
    // pre-pack W into B-fragment layout (two ds_load_b128 per fragment later)
    for (int p = tid; p < 2048; p += 256) {
        int i    = p & 7;
        int ln   = (p >> 3) & 31;
        int t    = (p >> 8) & 3;
        int kki  = p >> 10;
        int n    = t * 16 + (ln & 15);
        int k    = kki * 32 + (ln >> 4) * 8 + (i >> 2) * 16 + 2 * (i & 3);
        lwp[p] = pack2bf(Wg[k * HDIM + n], Wg[(k + 1) * HDIM + n]);
    }
    if (MODE == 0 && tid < HDIM) { lsum[tid] = 0.f; lsq[tid] = 0.f; }
    if (MODE == 1 && tid < 2 * HDIM) lss[tid] = ss[tid];
    __syncthreads();

    int wave = tid >> 5, lane = tid & 31;
    int half = lane >> 4, l16 = lane & 15;
    int m0   = blockIdx.x * 128 + wave * 16;
    int mrow = m0 + l16;
    int mclamp = mrow < N ? mrow : (N - 1);   // clamp: rows are independent
    const float4* ap4 = (const float4*)(A + (size_t)mclamp * HDIM);

    const v8u* lwv = (const v8u*)lwp;
    v8f c[4] = {v8f{}, v8f{}, v8f{}, v8f{}};

    #pragma unroll
    for (int kk = 0; kk < HDIM; kk += 32) {
        // ---- A fragment: 16x32 bf16; unconditional float4 loads ----
        int kb0 = kk + half * 8;        // dwords 0..3  -> K kb0..kb0+7
        int kb1 = kk + 16 + half * 8;   // dwords 4..7  -> K kb1..kb1+7
        float4 fa0 = ap4[(kb0 >> 2) + 0];
        float4 fa1 = ap4[(kb0 >> 2) + 1];
        float4 fb0 = ap4[(kb1 >> 2) + 0];
        float4 fb1 = ap4[(kb1 >> 2) + 1];
        if (MODE == 1) {
            float* va[4] = {&fa0.x, &fa1.x, &fb0.x, &fb1.x};
            int    kb[4] = {kb0, kb0 + 4, kb1, kb1 + 4};
            #pragma unroll
            for (int v = 0; v < 4; ++v)
                #pragma unroll
                for (int j = 0; j < 4; ++j)
                    va[v][j] = fmaxf(fmaf(va[v][j], lss[kb[v] + j], lss[HDIM + kb[v] + j]), 0.f);
        }
        v8u pa;
        pa[0] = pack2bf(fa0.x, fa0.y);  pa[1] = pack2bf(fa0.z, fa0.w);
        pa[2] = pack2bf(fa1.x, fa1.y);  pa[3] = pack2bf(fa1.z, fa1.w);
        pa[4] = pack2bf(fb0.x, fb0.y);  pa[5] = pack2bf(fb0.z, fb0.w);
        pa[6] = pack2bf(fb1.x, fb1.y);  pa[7] = pack2bf(fb1.z, fb1.w);
        v16bf af = __builtin_bit_cast(v16bf, pa);

        // ---- B fragments (pre-packed) + WMMA ----
        int kki = kk >> 5;
        #pragma unroll
        for (int t = 0; t < 4; ++t) {
            v8u pb = lwv[(kki * 4 + t) * 32 + lane];
            v16bf bf = __builtin_bit_cast(v16bf, pb);
            c[t] = __builtin_amdgcn_wmma_f32_16x16x32_bf16(
                false, af, false, bf, (short)0, c[t], false, false);
        }
    }

    // ---- epilogue phase 1: unguarded transform (single static erf copy) ----
    #pragma unroll
    for (int t = 0; t < 4; ++t) {
        float bv = bias[t * 16 + l16];
        #pragma unroll
        for (int j = 0; j < 8; ++j) {
            float z = c[t][j] + bv;
            if (MODE == 1) z = 0.5f * z * (1.f + erff(z * 0.70710678118654752f));
            c[t][j] = z;
        }
    }

    // ---- epilogue phase 2: stores (+stats); only cheap ops duplicated ----
    int rbase = m0 + half * 8;
    if (m0 + 16 <= N) {
        #pragma unroll
        for (int t = 0; t < 4; ++t) {
            int col = t * 16 + l16;
            float s = 0.f, s2 = 0.f;
            #pragma unroll
            for (int j = 0; j < 8; ++j) {
                float z = c[t][j];
                Out[(size_t)(rbase + j) * HDIM + col] = z;
                if (MODE == 0) { s += z; s2 += z * z; }
            }
            if (MODE == 0) { atomicAdd(&lsum[col], s); atomicAdd(&lsq[col], s2); }
        }
    } else {
        #pragma unroll
        for (int t = 0; t < 4; ++t) {
            int col = t * 16 + l16;
            float s = 0.f, s2 = 0.f;
            #pragma unroll
            for (int j = 0; j < 8; ++j) {
                if (rbase + j < N) {
                    float z = c[t][j];
                    Out[(size_t)(rbase + j) * HDIM + col] = z;
                    if (MODE == 0) { s += z; s2 += z * z; }
                }
            }
            if (MODE == 0) { atomicAdd(&lsum[col], s); atomicAdd(&lsq[col], s2); }
        }
    }
    if (MODE == 0) {
        __syncthreads();
        if (tid < HDIM) {
            atomicAdd(&stats[tid], lsum[tid]);
            atomicAdd(&stats[HDIM + tid], lsq[tid]);
        }
    }
}

// ---------------- launch ----------------

extern "C" void kernel_launch(void* const* d_in, const int* in_sizes, int n_in,
                              void* d_out, int out_size, void* d_ws, size_t ws_size,
                              hipStream_t stream) {
    const float* x      = (const float*)d_in[0];
    const int*   ei     = (const int*)  d_in[1];
    const int*   batch  = (const int*)  d_in[2];
    const float* cW1    = (const float*)d_in[4];
    const float* cb1    = (const float*)d_in[5];
    const float* cgam   = (const float*)d_in[6];
    const float* cbet   = (const float*)d_in[7];
    const float* cW2    = (const float*)d_in[8];
    const float* cb2    = (const float*)d_in[9];
    const float* hW1    = (const float*)d_in[10];
    const float* hb1    = (const float*)d_in[11];
    const float* hgam   = (const float*)d_in[12];
    const float* hbet   = (const float*)d_in[13];
    const float* hW2    = (const float*)d_in[14];
    const float* hb2    = (const float*)d_in[15];

    const int OUT = 2;
    int N = in_sizes[0] / HDIM;
    int E = in_sizes[1] / 2;
    int L = in_sizes[4] / (HDIM * HDIM);
    int G = out_size / OUT;

    size_t NH = (size_t)N * HDIM;
    size_t GH = (size_t)G * HDIM;
    float* ws    = (float*)d_ws;
    float* a     = ws;                 // [N,64] h + agg
    float* z1    = a + NH;             // [N,64] first-linear output
    float* hcur  = z1 + NH;            // [N,64] layer output
    float* stats = hcur + NH;          // [128]
    float* ssbuf = stats + 128;        // [128] scale/shift
    float* g     = ssbuf + 128;        // [G,64] pooled
    float* gz1   = g + GH;             // [G,64]

    int gemmBlocksN = (N + 127) / 128;
    int gemmBlocksG = (G + 127) / 128;
    int copyB  = (int)((NH / 4 + 255) / 256);
    int scatB  = (E * (HDIM / 4) + 255) / 256;
    int poolB  = (N * (HDIM / 4) + 255) / 256;

    const float* hin = x;
    for (int l = 0; l < L; ++l) {
        // a = h_in ; a[dst] += h_in[src]
        copy_f32x4<<<copyB, 256, 0, stream>>>((float4*)a, (const float4*)hin, (int)(NH / 4));
        scatter_add<<<scatB, 256, 0, stream>>>(hin, ei, a, E);
        // z1 = a @ W1 + b1, BN stats
        zero_f32<<<1, 128, 0, stream>>>(stats, 128);
        gin_gemm<0><<<gemmBlocksN, 256, 0, stream>>>(
            a, cW1 + (size_t)l * HDIM * HDIM, cb1 + l * HDIM, nullptr, z1, stats, N);
        bn_finalize<<<1, 64, 0, stream>>>(stats, cgam + l * HDIM, cbet + l * HDIM, ssbuf, (float)N);
        // h = gelu( relu(bn(z1)) @ W2 + b2 )
        gin_gemm<1><<<gemmBlocksN, 256, 0, stream>>>(
            z1, cW2 + (size_t)l * HDIM * HDIM, cb2 + l * HDIM, ssbuf, hcur, nullptr, N);
        hin = hcur;
    }

    // global_add_pool
    zero_f32<<<(int)((GH + 255) / 256), 256, 0, stream>>>(g, (int)GH);
    pool_add<<<poolB, 256, 0, stream>>>(hin, batch, g, N);

    // head: gz1 = g @ hW1 + hb1, BN stats over G rows
    zero_f32<<<1, 128, 0, stream>>>(stats, 128);
    gin_gemm<0><<<gemmBlocksG, 256, 0, stream>>>(g, hW1, hb1, nullptr, gz1, stats, G);
    bn_finalize<<<1, 64, 0, stream>>>(stats, hgam, hbet, ssbuf, (float)G);

    // out = relu(bn(gz1)) @ hW2 + hb2
    head_out_kernel<<<(G * OUT + 255) / 256, 256, 0, stream>>>(
        gz1, ssbuf, hW2, hb2, (float*)d_out, G, OUT);
}
